// TransformerDecoderBlock_11519102288117
// MI455X (gfx1250) — compile-verified, matrix operand order checked by reading, not run on hardware
//
#include <hip/hip_runtime.h>
#include <cstdint>
#include <cstddef>

// ---------------------------------------------------------------------------
// MI455X (gfx1250) transformer decoder block.
// Dense math through v_wmma_f32_16x16x32_bf16 (wave32), fp32 accumulate.
// GEMM tiles staged via double-buffered global_load_async_to_lds_b128
// (ASYNCcnt path), overlapping DMA with WMMA.
// LN / softmax / residual / GELU in fp32. bf16 stored as raw u16.
// ---------------------------------------------------------------------------

typedef unsigned short u16t;
typedef unsigned int   u32t;
typedef __attribute__((ext_vector_type(16))) __bf16 v16bf;
typedef __attribute__((ext_vector_type(8)))  float  v8f;

#define CEMB   768
#define NHEAD  8
#define HDIM   96
#define NEG_BIG (-3.402823466e38f)
#define NEG_INF (-__builtin_huge_valf())

__device__ __forceinline__ u16t f2bf(float f){
  u32t u = __float_as_uint(f);
  if ((u & 0x7F800000u) == 0x7F800000u) return (u16t)(u >> 16);   // Inf/NaN
  u32t r = u + 0x7FFFu + ((u >> 16) & 1u);                        // RNE
  return (u16t)(r >> 16);
}

// CDNA5 16-bit A-matrix K layout: dword v (0..7), lane-half g (0/1) holds
// K = {base, base+1}, base = v<4 ? 2v+8g : 16+2(v-4)+8g   (ISA 7.12.2)
__device__ __forceinline__ int kpair(int v, int g){
  return (v < 4) ? (2*v + 8*g) : (16 + 2*(v-4) + 8*g);
}

union Frag {
  u16t u[16];
  u32t d[8];
  v16bf v;
};

__device__ __forceinline__ v8f wmma_bf16(v16bf a, v16bf b, v8f c){
  return __builtin_amdgcn_wmma_f32_16x16x32_bf16(false, a, false, b,
                                                 (short)0, c, false, false);
}

__device__ __forceinline__ v8f vzero8(){
  v8f z = {0.f,0.f,0.f,0.f,0.f,0.f,0.f,0.f};
  return z;
}

// Per-lane async DMA of 16 bytes global -> LDS, tracked with ASYNCcnt.
__device__ __forceinline__ void async_g2l_b128(unsigned lds_off, const u16t* g){
  unsigned long long ga = (unsigned long long)g;
  asm volatile("global_load_async_to_lds_b128 %0, %1, off"
               :: "v"(lds_off), "v"(ga) : "memory");
}
__device__ __forceinline__ void wait_async0(){
  asm volatile("s_wait_asynccnt 0x0" ::: "memory");
}
// Low 32 bits of a flat LDS address are the LDS byte offset (ISA 10.2).
__device__ __forceinline__ unsigned lds_off32(const void* p){
  return (unsigned)(uintptr_t)p;
}

// ---------------------------------------------------------------------------
// fp32 -> bf16 elementwise (weight conversion)
// ---------------------------------------------------------------------------
__global__ void cvt_bf16(const float* __restrict__ in, u16t* __restrict__ out, int n){
  int i = blockIdx.x * blockDim.x + threadIdx.x;
  if (i < n) out[i] = f2bf(in[i]);
}

// ---------------------------------------------------------------------------
// LayerNorm over C=768 (one block of 256 threads per row), bf16 output.
// jnp.var is population variance; eps = 1e-5.
// ---------------------------------------------------------------------------
__global__ __launch_bounds__(256) void ln768_bf16(
    const float* __restrict__ in, u16t* __restrict__ out,
    const float* __restrict__ gamma, const float* __restrict__ beta){
  const int row = blockIdx.x;
  const int t = threadIdx.x;
  const float* p = in + (size_t)row * CEMB;
  float x0 = p[t], x1 = p[t+256], x2 = p[t+512];
  float s  = x0 + x1 + x2;
  float ss = x0*x0 + x1*x1 + x2*x2;
  #pragma unroll
  for (int off = 16; off >= 1; off >>= 1){
    s  += __shfl_xor(s,  off, 32);
    ss += __shfl_xor(ss, off, 32);
  }
  __shared__ float shs[8], shss[8];
  const int wid = t >> 5;
  if ((t & 31) == 0){ shs[wid] = s; shss[wid] = ss; }
  __syncthreads();
  s = 0.f; ss = 0.f;
  #pragma unroll
  for (int i = 0; i < 8; ++i){ s += shs[i]; ss += shss[i]; }
  const float mean = s * (1.0f / CEMB);
  const float var  = ss * (1.0f / CEMB) - mean * mean;
  const float rs   = rsqrtf(var + 1e-5f);
  u16t* q = out + (size_t)row * CEMB;
  q[t]     = f2bf((x0 - mean) * rs * gamma[t]     + beta[t]);
  q[t+256] = f2bf((x1 - mean) * rs * gamma[t+256] + beta[t+256]);
  q[t+512] = f2bf((x2 - mean) * rs * gamma[t+512] + beta[t+512]);
}

// ---------------------------------------------------------------------------
// bf16 WMMA GEMM: C[M,N] = A[M,K] * B[K,N] (+bias, +GELU or +residual).
// Block tile 128x64, BK=32, 8 waves (4x2), each wave a 32x32 tile (4 WMMAs).
// Tiles staged with double-buffered async global->LDS DMA (ASYNCcnt).
// Requires M%128==0, N%64==0, K%32==0 (true for all call sites here).
// MODE 0: bf16 out. MODE 1: bf16 out with exact GELU. MODE 2: f32 out =
//         residual + acc + bias.
// ---------------------------------------------------------------------------
template<int MODE>
__global__ __launch_bounds__(256) void gemm_bf16_wmma(
    const u16t* __restrict__ A, const u16t* __restrict__ Bw,
    const float* __restrict__ bias, const float* __restrict__ res,
    void* __restrict__ out, int M, int N, int K){
  __shared__ alignas(16) u16t As[2][128 * 32];   // 2 x 8 KB
  __shared__ alignas(16) u16t Bs[2][32 * 64];    // 2 x 4 KB
  const int tid  = threadIdx.x;
  const int wave = tid >> 5, lane = tid & 31;
  const int nl = lane & 15, lg = lane >> 4;
  const int wm = wave & 3,  wn = wave >> 2;
  const int m0 = blockIdx.y * 128, n0 = blockIdx.x * 64;
  (void)M;
  v8f acc[2][2];
  acc[0][0] = vzero8(); acc[0][1] = vzero8();
  acc[1][0] = vzero8(); acc[1][1] = vzero8();

  // A tile: 512 x 16B chunks (2/thread); B tile: 256 x 16B chunks (1/thread)
  auto issue_tile = [&](int k0, int buf){
    unsigned la = lds_off32(&As[buf][0]);
    unsigned lb = lds_off32(&Bs[buf][0]);
    #pragma unroll
    for (int i = 0; i < 2; ++i){
      int idx4 = tid + i * 256;
      int row = idx4 >> 2, c4 = idx4 & 3;
      async_g2l_b128(la + (unsigned)(row * 64 + c4 * 16),
                     A + (size_t)(m0 + row) * K + k0 + c4 * 8);
    }
    {
      int row = tid >> 3, c4 = tid & 7;
      async_g2l_b128(lb + (unsigned)(row * 128 + c4 * 16),
                     Bw + (size_t)(k0 + row) * N + n0 + c4 * 8);
    }
  };

  const int T = K >> 5;
  issue_tile(0, 0);
  for (int t = 0; t < T; ++t){
    wait_async0();        // this wave's tile-t transfers landed in LDS
    __syncthreads();      // everyone's tile-t visible; prior reads retired
    if (t + 1 < T) issue_tile((t + 1) << 5, (t + 1) & 1);  // overlap with compute

    const u16t* Asb = &As[t & 1][0];
    const u16t* Bsb = &Bs[t & 1][0];
    Frag af[2], bfr[2];
    #pragma unroll
    for (int mt = 0; mt < 2; ++mt){
      int arow = (wm * 32 + mt * 16 + nl) * 32;
      #pragma unroll
      for (int v = 0; v < 8; ++v)
        af[mt].d[v] = *(const u32t*)(Asb + arow + kpair(v, lg));
    }
    #pragma unroll
    for (int nt = 0; nt < 2; ++nt){
      int bcol = wn * 32 + nt * 16 + nl;
      #pragma unroll
      for (int e = 0; e < 16; ++e){
        int kk = kpair(e >> 1, lg) + (e & 1);
        bfr[nt].u[e] = Bsb[kk * 64 + bcol];
      }
    }
    #pragma unroll
    for (int mt = 0; mt < 2; ++mt)
      #pragma unroll
      for (int nt = 0; nt < 2; ++nt)
        acc[mt][nt] = wmma_bf16(af[mt].v, bfr[nt].v, acc[mt][nt]);
  }

  // Epilogue. C/D layout: vgpr r -> row r + 8*lg, lane nl -> col.
  #pragma unroll
  for (int mt = 0; mt < 2; ++mt){
    #pragma unroll
    for (int nt = 0; nt < 2; ++nt){
      int gn = n0 + wn * 32 + nt * 16 + nl;
      float bv = bias ? bias[gn] : 0.0f;
      #pragma unroll
      for (int r = 0; r < 8; ++r){
        int gm = m0 + wm * 32 + mt * 16 + r + 8 * lg;
        float vv = acc[mt][nt][r] + bv;
        size_t o = (size_t)gm * N + gn;
        if (MODE == 0){
          ((u16t*)out)[o] = f2bf(vv);
        } else if (MODE == 1){
          vv = 0.5f * vv * (1.0f + erff(vv * 0.70710678118654752440f)); // exact GELU
          ((u16t*)out)[o] = f2bf(vv);
        } else {
          ((float*)out)[o] = res[o] + vv;
        }
      }
    }
  }
}

// ---------------------------------------------------------------------------
// Flash attention, one wave per (b, h, 16-row i-tile); 4 waves / block.
// q: [B*L, 768] bf16 (head h at cols h*96..). kv: [B*Lk, 1536] bf16
// (k at cols 0..767, v at 768..1535). Ragged masks per reference:
//   keep = (i < eff[b]*us) && (j < (self ? eff[b]*us : eff[b]))
// masked entries -> -FLT_MAX before softmax (fully-masked rows degrade to
// uniform attention, matching the reference).
// ---------------------------------------------------------------------------
__global__ __launch_bounds__(128) void flash_attn_wmma(
    const u16t* __restrict__ qbuf, const u16t* __restrict__ kvbuf,
    u16t* __restrict__ obuf, const int* __restrict__ eff,
    const int* __restrict__ usp, int L, int Lk, int j_uses_us){
  __shared__ alignas(16) u16t Pl[4 * 16 * 32];   // per-wave P tile (bf16)
  const int lane = threadIdx.x & 31, wave = threadIdx.x >> 5;
  const int nl = lane & 15, lg = lane >> 4;
  const int b = blockIdx.z, h = blockIdx.y;
  const int i0 = (blockIdx.x * 4 + wave) * 16;
  const int us = usp[0];
  const int el_i = eff[b] * us;
  const int el_j = j_uses_us ? el_i : eff[b];
  const float scale = 0.10206207261596576f;  // 96^-0.5

  // Q fragments: A-layout, d=96 -> 3 chunks of K=32
  Frag qf[3];
  {
    size_t qrow = ((size_t)(b * L + i0 + nl)) * CEMB + h * HDIM;
    #pragma unroll
    for (int c = 0; c < 3; ++c)
      #pragma unroll
      for (int v = 0; v < 8; ++v)
        qf[c].d[v] = *(const u32t*)(qbuf + qrow + c * 32 + kpair(v, lg));
  }

  float mrow[8], lrow[8];
  v8f oacc[6];
  #pragma unroll
  for (int r = 0; r < 8; ++r){ mrow[r] = NEG_INF; lrow[r] = 0.0f; }
  #pragma unroll
  for (int dc = 0; dc < 6; ++dc) oacc[dc] = vzero8();

  for (int j0 = 0; j0 < Lk; j0 += 32){
    // ---- scores for two 16-j subtiles -----------------------------------
    v8f sacc[2]; sacc[0] = vzero8(); sacc[1] = vzero8();
    #pragma unroll
    for (int jt = 0; jt < 2; ++jt){
      int j = j0 + jt * 16 + nl;
      size_t krow = ((size_t)(b * Lk + j)) * (2 * CEMB) + h * HDIM;
      #pragma unroll
      for (int c = 0; c < 3; ++c){
        Frag kf;
        #pragma unroll
        for (int v = 0; v < 8; ++v)
          kf.d[v] = *(const u32t*)(kvbuf + krow + c * 32 + kpair(v, lg));
        sacc[jt] = wmma_bf16(qf[c].v, kf.v, sacc[jt]);
      }
    }
    // ---- mask + online softmax ------------------------------------------
    float p0[8], p1[8], tmax[8];
    const int jA = j0 + nl, jB = j0 + 16 + nl;
    #pragma unroll
    for (int r = 0; r < 8; ++r){
      int irow = i0 + r + 8 * lg;
      bool rok = irow < el_i;
      float s0 = sacc[0][r] * scale;  s0 = (rok && jA < el_j) ? s0 : NEG_BIG;
      float s1 = sacc[1][r] * scale;  s1 = (rok && jB < el_j) ? s1 : NEG_BIG;
      p0[r] = s0; p1[r] = s1;
      tmax[r] = fmaxf(s0, s1);
    }
    #pragma unroll
    for (int r = 0; r < 8; ++r)
      #pragma unroll
      for (int off = 8; off >= 1; off >>= 1)
        tmax[r] = fmaxf(tmax[r], __shfl_xor(tmax[r], off, 32));
    #pragma unroll
    for (int r = 0; r < 8; ++r){
      float mn = fmaxf(mrow[r], tmax[r]);
      float alpha = __expf(mrow[r] - mn);
      mrow[r] = mn;
      float e0 = __expf(p0[r] - mn);
      float e1 = __expf(p1[r] - mn);
      float rs = e0 + e1;
      #pragma unroll
      for (int off = 8; off >= 1; off >>= 1) rs += __shfl_xor(rs, off, 32);
      lrow[r] = lrow[r] * alpha + rs;
      #pragma unroll
      for (int dc = 0; dc < 6; ++dc) oacc[dc][r] *= alpha;
      int m = r + 8 * lg;
      Pl[wave * 512 + m * 32 + nl]      = f2bf(e0);   // D-layout -> LDS tile
      Pl[wave * 512 + m * 32 + 16 + nl] = f2bf(e1);
    }
    __syncthreads();
    // ---- re-layout P into an A fragment [16 x 32] -----------------------
    Frag pa;
    {
      int prow = wave * 512 + nl * 32;
      #pragma unroll
      for (int v = 0; v < 8; ++v)
        pa.d[v] = *(const u32t*)(Pl + prow + kpair(v, lg));
    }
    __syncthreads();
    // ---- out += P @ V ----------------------------------------------------
    #pragma unroll
    for (int dc = 0; dc < 6; ++dc){
      Frag vf;
      #pragma unroll
      for (int e = 0; e < 16; ++e){
        int jj = kpair(e >> 1, lg) + (e & 1);
        vf.u[e] = kvbuf[((size_t)(b * Lk + j0 + jj)) * (2 * CEMB)
                        + CEMB + h * HDIM + dc * 16 + nl];
      }
      oacc[dc] = wmma_bf16(pa.v, vf.v, oacc[dc]);
    }
  }
  // ---- normalize + store bf16 -------------------------------------------
  #pragma unroll
  for (int dc = 0; dc < 6; ++dc){
    #pragma unroll
    for (int r = 0; r < 8; ++r){
      int irow = i0 + r + 8 * lg;
      float val = oacc[dc][r] / lrow[r];
      obuf[((size_t)(b * L + irow)) * CEMB + h * HDIM + dc * 16 + nl] = f2bf(val);
    }
  }
}

// ---------------------------------------------------------------------------
// Host orchestration
// ---------------------------------------------------------------------------
extern "C" void kernel_launch(void* const* d_in, const int* in_sizes, int n_in,
                              void* d_out, int out_size, void* d_ws, size_t ws_size,
                              hipStream_t stream){
  (void)n_in; (void)out_size; (void)ws_size;
  const float* x    = (const float*)d_in[0];
  const float* cctx = (const float*)d_in[1];
  const int*   eff  = (const int*)d_in[2];
  const int*   usp  = (const int*)d_in[3];
  const float* Wq1  = (const float*)d_in[4];
  const float* Wkv1 = (const float*)d_in[5];
  const float* Wo1  = (const float*)d_in[6];
  const float* bo1  = (const float*)d_in[7];
  const float* Wq2  = (const float*)d_in[8];
  const float* Wkv2 = (const float*)d_in[9];
  const float* Wo2  = (const float*)d_in[10];
  const float* bo2  = (const float*)d_in[11];
  const float* g1   = (const float*)d_in[12];
  const float* b1   = (const float*)d_in[13];
  const float* g2   = (const float*)d_in[14];
  const float* b2   = (const float*)d_in[15];
  const float* g4   = (const float*)d_in[16];
  const float* b4   = (const float*)d_in[17];
  const float* gc   = (const float*)d_in[18];
  const float* bc   = (const float*)d_in[19];
  const float* Wf1  = (const float*)d_in[20];
  const float* bf1  = (const float*)d_in[21];
  const float* Wf2  = (const float*)d_in[22];
  const float* bf2  = (const float*)d_in[23];

  const int Cc = CEMB;
  const int B  = in_sizes[2];
  const int L  = in_sizes[0] / (B * Cc);   // 1024
  const int S  = in_sizes[1] / (B * Cc);   // 512
  const int F  = in_sizes[21];             // 3072
  const int BL = B * L;                    // 8192
  const int BS = B * S;                    // 4096

  // --- workspace carve-up -------------------------------------------------
  char* wp = (char*)d_ws;
  auto alloc = [&](size_t bytes) -> void* {
    void* p = wp; wp += (bytes + 255) & ~(size_t)255; return p;
  };
  float* x_cur = (float*)alloc((size_t)BL * Cc * 4);
  u16t*  xn    = (u16t*)alloc((size_t)BL * Cc * 2);
  u16t*  cn    = (u16t*)alloc((size_t)BS * Cc * 2);
  u16t*  qb    = (u16t*)alloc((size_t)BL * Cc * 2);
  u16t*  kvb   = (u16t*)alloc((size_t)BL * 2 * Cc * 2);
  u16t*  ab    = (u16t*)alloc((size_t)BL * Cc * 2);
  u16t*  midb  = (u16t*)alloc((size_t)BL * F * 2);
  u16t*  wq1b  = (u16t*)alloc((size_t)Cc * Cc * 2);
  u16t*  wkv1b = (u16t*)alloc((size_t)Cc * 2 * Cc * 2);
  u16t*  wo1b  = (u16t*)alloc((size_t)Cc * Cc * 2);
  u16t*  wq2b  = (u16t*)alloc((size_t)Cc * Cc * 2);
  u16t*  wkv2b = (u16t*)alloc((size_t)Cc * 2 * Cc * 2);
  u16t*  wo2b  = (u16t*)alloc((size_t)Cc * Cc * 2);
  u16t*  wf1b  = (u16t*)alloc((size_t)Cc * F * 2);
  u16t*  wf2b  = (u16t*)alloc((size_t)F * Cc * 2);

  auto cvt = [&](const float* src, u16t* dst, size_t n){
    cvt_bf16<<<dim3((unsigned)((n + 255) / 256)), dim3(256), 0, stream>>>(src, dst, (int)n);
  };
  cvt(Wq1,  wq1b,  (size_t)Cc * Cc);
  cvt(Wkv1, wkv1b, (size_t)Cc * 2 * Cc);
  cvt(Wo1,  wo1b,  (size_t)Cc * Cc);
  cvt(Wq2,  wq2b,  (size_t)Cc * Cc);
  cvt(Wkv2, wkv2b, (size_t)Cc * 2 * Cc);
  cvt(Wo2,  wo2b,  (size_t)Cc * Cc);
  cvt(Wf1,  wf1b,  (size_t)Cc * F);
  cvt(Wf2,  wf2b,  (size_t)F * Cc);

  const dim3 blk256(256);
  auto gemm0 = [&](const u16t* A, const u16t* Bm, u16t* O, int M, int N, int K){
    gemm_bf16_wmma<0><<<dim3(N / 64, M / 128), blk256, 0, stream>>>(
        A, Bm, nullptr, nullptr, (void*)O, M, N, K);
  };
  auto gemm1 = [&](const u16t* A, const u16t* Bm, const float* bias, u16t* O,
                   int M, int N, int K){
    gemm_bf16_wmma<1><<<dim3(N / 64, M / 128), blk256, 0, stream>>>(
        A, Bm, bias, nullptr, (void*)O, M, N, K);
  };
  auto gemm2 = [&](const u16t* A, const u16t* Bm, const float* bias,
                   const float* res, float* O, int M, int N, int K){
    gemm_bf16_wmma<2><<<dim3(N / 64, M / 128), blk256, 0, stream>>>(
        A, Bm, bias, res, (void*)O, M, N, K);
  };

  // ---- x_cur = x ----------------------------------------------------------
  hipMemcpyAsync(x_cur, x, (size_t)BL * Cc * 4, hipMemcpyDeviceToDevice, stream);

  // ---- self attention -----------------------------------------------------
  ln768_bf16<<<dim3(BL), blk256, 0, stream>>>(x_cur, xn, g1, b1);
  gemm0(xn, wq1b,  qb,  BL, Cc,     Cc);
  gemm0(xn, wkv1b, kvb, BL, 2 * Cc, Cc);
  flash_attn_wmma<<<dim3(L / 64, NHEAD, B), dim3(128), 0, stream>>>(
      qb, kvb, ab, eff, usp, L, L, /*j_uses_us=*/1);
  gemm2(ab, wo1b, bo1, x_cur, x_cur, BL, Cc, Cc);   // x += attn @ Wo1 + bo1

  // ---- cross attention ----------------------------------------------------
  ln768_bf16<<<dim3(BL), blk256, 0, stream>>>(x_cur, xn, g2, b2);
  ln768_bf16<<<dim3(BS), blk256, 0, stream>>>(cctx, cn, gc, bc);
  gemm0(xn, wq2b,  qb,  BL, Cc,     Cc);
  gemm0(cn, wkv2b, kvb, BS, 2 * Cc, Cc);
  flash_attn_wmma<<<dim3(L / 64, NHEAD, B), dim3(128), 0, stream>>>(
      qb, kvb, ab, eff, usp, L, S, /*j_uses_us=*/0);
  gemm2(ab, wo2b, bo2, x_cur, x_cur, BL, Cc, Cc);   // x += attn @ Wo2 + bo2

  // ---- FFN ---------------------------------------------------------------
  ln768_bf16<<<dim3(BL), blk256, 0, stream>>>(x_cur, xn, g4, b4);
  gemm1(xn, wf1b, bf1, midb, BL, F, Cc);            // gelu(xn@Wf1 + bf1) -> bf16
  gemm2(midb, wf2b, bf2, x_cur, (float*)d_out, BL, Cc, F);  // out = x + mid@Wf2 + bf2
}
